// EchoStateNetwork_59682865545465
// MI455X (gfx1250) — matrix operand order repaired and checked by reading, not compiled, not run on hardware
//
#include <hip/hip_runtime.h>
#include <hip/hip_bf16.h>
#include <math.h>

// Problem constants (from reference)
#define NRES   2048
#define TLEN   4096
#define BATCH  16
#define NWG    16          // persistent workgroups; must be co-resident
#define THREADS 256        // 8 waves of 32
#define WAVES   (THREADS / 32)
#define COLS_PER_WG (NRES / NWG)   // 128 columns -> 8 waves x 16-col tile
#define LDS_ROW 2056       // padded bf16 row stride (2048 + 8) -> distinct LDS banks per row
#define SMEM_BYTES (BATCH * LDS_ROW * 2 + WAVES * 16 * 4)

typedef __attribute__((ext_vector_type(16))) __bf16 v16bf;
typedef __attribute__((ext_vector_type(8)))  __bf16 bf16x8;
typedef __attribute__((ext_vector_type(8)))  float  v8f;

// ---- workspace layout (bytes) ----
#define WS_WB       0                         // W_res in bf16: 2048*2048*2 = 8388608
#define WS_S0       (8388608)                 // state ping: 16*2048*2 = 65536
#define WS_S1       (WS_S0 + 65536)           // state pong
#define WS_SYNC     (WS_S1 + 65536)           // 2 uints (count, generation), 256B slot
#define WS_PART     (WS_SYNC + 256)           // partials: NWG*TLEN*16 f32 = 4194304
// total ~12.7 MB

// Branch-free tanh: single hardware transcendental if available, else exp2/rcp form.
__device__ __forceinline__ float fast_tanh(float x) {
#if __has_builtin(__builtin_amdgcn_tanhf)
  return __builtin_amdgcn_tanhf(x);
#else
  // tanh(x) = 1 - 2/(exp2(2x*log2e)+1); rcp(inf)=0 -> +/-1 at the tails.
  float t = __builtin_amdgcn_exp2f(x * 2.885390081777927f);
  return 1.0f - 2.0f * __builtin_amdgcn_rcpf(t + 1.0f);
#endif
}

__device__ __forceinline__ void grid_barrier(unsigned* cnt, unsigned* gen) {
  __syncthreads();
  if (threadIdx.x == 0) {
    __builtin_amdgcn_fence(__ATOMIC_RELEASE, "agent");   // publish state/partials
    unsigned g = __hip_atomic_load(gen, __ATOMIC_RELAXED, __HIP_MEMORY_SCOPE_AGENT);
    unsigned prev = __hip_atomic_fetch_add(cnt, 1u, __ATOMIC_RELAXED, __HIP_MEMORY_SCOPE_AGENT);
    if (prev == NWG - 1) {
      __hip_atomic_store(cnt, 0u, __ATOMIC_RELAXED, __HIP_MEMORY_SCOPE_AGENT);
      __hip_atomic_fetch_add(gen, 1u, __ATOMIC_RELEASE, __HIP_MEMORY_SCOPE_AGENT);
    } else {
      while (__hip_atomic_load(gen, __ATOMIC_RELAXED, __HIP_MEMORY_SCOPE_AGENT) == g)
        __builtin_amdgcn_s_sleep(2);
    }
    __builtin_amdgcn_fence(__ATOMIC_ACQUIRE, "agent");   // see others' state
  }
  __syncthreads();
}

// ---- prep: W_res f32 -> bf16 (L2-resident 8MB), zero initial state + sync ----
__global__ void esn_prep(const float* __restrict__ W_res, __bf16* __restrict__ Wb,
                         unsigned* __restrict__ sync, __bf16* __restrict__ S0) {
  int idx = blockIdx.x * blockDim.x + threadIdx.x;
  if (idx < NRES * NRES) Wb[idx] = (__bf16)W_res[idx];
  if (idx < BATCH * NRES) S0[idx] = (__bf16)0.0f;
  if (idx < 2) sync[idx] = 0u;
}

// ---- main persistent recurrence kernel ----
__global__ __launch_bounds__(THREADS, 1)
void esn_main(const float* __restrict__ X,      // [16][4096]
              const float* __restrict__ W_in,   // [2048]
              const float* __restrict__ W_out,  // [2049]
              const __bf16* __restrict__ Wb,    // [2048][2048] row-major (n,k)
              __bf16* __restrict__ S0,
              __bf16* __restrict__ S1,
              unsigned* __restrict__ sync,
              float* __restrict__ partials)     // [NWG][TLEN][16]
{
  extern __shared__ char smem[];
  __bf16* lds_state = (__bf16*)smem;                               // [16][LDS_ROW]
  float*  lds_part  = (float*)(smem + BATCH * LDS_ROW * 2);        // [WAVES][16]

  const int tid  = threadIdx.x;
  const int wg   = blockIdx.x;
  const int lane = tid & 31;
  const int wave = tid >> 5;
  const int h    = lane >> 4;        // K-half selector of WMMA fragment layout
  const int m16  = lane & 15;
  const int ncol = wg * COLS_PER_WG + wave * 16 + m16;   // this lane's output column n

  const float w_in_n  = W_in[ncol];
  const float w_out_n = W_out[ncol];
  const __bf16* Brow  = Wb + (size_t)ncol * NRES;

  v8f s_old = {};          // f32 state slice, C-fragment layout: row b = r + 8h, col = ncol
  __bf16* Scur = S0;
  __bf16* Snxt = S1;

  for (int t = 0; t < TLEN; ++t) {
    // stage current bf16 state [16][2048] into padded LDS
    {
      const uint4* src = (const uint4*)Scur;
      #pragma unroll
      for (int i = 0; i < 16; ++i) {
        int idx  = i * THREADS + tid;     // 4096 uint4 total
        int row  = idx >> 8;              // / 256 uint4 per row
        int col  = idx & 255;
        *(uint4*)(lds_state + row * LDS_ROW + col * 8) = src[idx];
      }
    }
    __syncthreads();

    // pre[b, ncol] = sum_k s[b,k] * W_res[ncol,k]  via 64 x v_wmma_f32_16x16x32_bf16
    v8f acc = {};
    const __bf16* Arow = lds_state + m16 * LDS_ROW;
    #pragma unroll 8
    for (int k0 = 0; k0 < NRES; k0 += 32) {
      bf16x8 a0 = *(const bf16x8*)(Arow + k0 + 8 * h);        // K = k0+8h .. +7
      bf16x8 a1 = *(const bf16x8*)(Arow + k0 + 16 + 8 * h);   // K = k0+16+8h .. +7
      bf16x8 b0 = *(const bf16x8*)(Brow + k0 + 8 * h);        // K = k0+8h .. +7
      bf16x8 b1 = *(const bf16x8*)(Brow + k0 + 16 + 8 * h);   // K = k0+16+8h .. +7
      v16bf a, b;
      #pragma unroll
      for (int i = 0; i < 8; ++i) {
        a[i] = a0[i]; a[i + 8] = a1[i];
        b[i] = b0[i]; b[i + 8] = b1[i];
      }
      acc = __builtin_amdgcn_wmma_f32_16x16x32_bf16(
          /*neg_a=*/false, a, /*neg_b=*/false, b,
          /*c_mod=*/(short)0, acc, /*reuse_a=*/false, /*reuse_b=*/false);
    }

    // input term + tanh + leaky update; publish bf16 state; readout partials
    float p[8];
    #pragma unroll
    for (int r = 0; r < 8; ++r) {
      const int brow = r + 8 * h;
      float pre = acc[r] + X[brow * TLEN + t] * w_in_n;
      float sn  = 0.99f * fast_tanh(pre) + 0.01f * s_old[r];
      s_old[r]  = sn;
      Snxt[(size_t)brow * NRES + ncol] = (__bf16)sn;
      p[r] = w_out_n * sn;
    }
    // reduce W_out·s over the 16 lanes of each K-half (masks stay inside group)
    #pragma unroll
    for (int mask = 1; mask < 16; mask <<= 1) {
      #pragma unroll
      for (int r = 0; r < 8; ++r) p[r] += __shfl_xor(p[r], mask, 32);
    }
    if (m16 == 0) {
      #pragma unroll
      for (int r = 0; r < 8; ++r) lds_part[wave * 16 + r + 8 * h] = p[r];
    }
    __syncthreads();
    if (tid < 16) {
      float sum = 0.0f;
      #pragma unroll
      for (int w = 0; w < WAVES; ++w) sum += lds_part[w * 16 + tid];
      partials[((size_t)wg * TLEN + t) * 16 + tid] = sum;
    }

    grid_barrier(sync, sync + 1);   // all writes of step t visible everywhere
    __bf16* tmp = Scur; Scur = Snxt; Snxt = tmp;
  }
}

// ---- finish: out[b,t] = b_out + W_out[N]*x[b,t] + sum_wg partials[wg,t,b] ----
__global__ void esn_finish(const float* __restrict__ X, const float* __restrict__ W_out,
                           const float* __restrict__ b_out,
                           const float* __restrict__ partials, float* __restrict__ out) {
  int idx = blockIdx.x * blockDim.x + threadIdx.x;   // 0 .. 16*4096-1
  if (idx >= BATCH * TLEN) return;
  int b = idx / TLEN;
  int t = idx - b * TLEN;
  float acc = b_out[0] + W_out[NRES] * X[idx];
  #pragma unroll 8
  for (int wg = 0; wg < NWG; ++wg)
    acc += partials[((size_t)wg * TLEN + t) * 16 + b];
  out[idx] = acc;
}

extern "C" void kernel_launch(void* const* d_in, const int* in_sizes, int n_in,
                              void* d_out, int out_size, void* d_ws, size_t ws_size,
                              hipStream_t stream) {
  const float* X     = (const float*)d_in[0];   // [16][4096][1]
  const float* W_in  = (const float*)d_in[1];   // [2048][1]
  const float* W_res = (const float*)d_in[2];   // [2048][2048]
  const float* W_out = (const float*)d_in[3];   // [1][2049]
  const float* b_out = (const float*)d_in[4];   // [1]
  float* out = (float*)d_out;

  char* ws = (char*)d_ws;
  __bf16*   Wb       = (__bf16*)(ws + WS_WB);
  __bf16*   S0       = (__bf16*)(ws + WS_S0);
  __bf16*   S1       = (__bf16*)(ws + WS_S1);
  unsigned* sync     = (unsigned*)(ws + WS_SYNC);
  float*    partials = (float*)(ws + WS_PART);

  (void)in_sizes; (void)n_in; (void)out_size; (void)ws_size;

  hipFuncSetAttribute((const void*)esn_main,
                      hipFuncAttributeMaxDynamicSharedMemorySize, (int)SMEM_BYTES);

  esn_prep<<<(NRES * NRES + 255) / 256, 256, 0, stream>>>(W_res, Wb, sync, S0);
  esn_main<<<NWG, THREADS, SMEM_BYTES, stream>>>(X, W_in, W_out, Wb, S0, S1, sync, partials);
  esn_finish<<<(BATCH * TLEN + 255) / 256, 256, 0, stream>>>(X, W_out, b_out, partials, out);
}
